// Trajectory_SelfAttention_24060406792682
// MI455X (gfx1250) — matrix operand order repaired and checked by reading, not compiled, not run on hardware
//
#include <hip/hip_runtime.h>

// ---------------- problem constants ----------------
#define BB   32
#define DDm  512
#define TTm  32
#define NNm  64
#define HHm  8
#define DKm  64                    // DDm / HHm
#define TDm  (TTm * DDm)           // 16384 : stride between nodes n in (B,N,T,D)
#define NTD  (NNm * TTm * DDm)     // 1048576 : per-batch layernorm group
#define MROWS (BB * NNm * TTm)     // 65536 GEMM rows
#define FDIM (TTm * DKm)           // 2048 per-head feature dim
#define ATT_SCALE 0.022097086912079610f   // (64*32)^-0.5
#define LN_EPS 1e-5f

typedef __attribute__((ext_vector_type(16))) _Float16 v16h;
typedef __attribute__((ext_vector_type(8)))  float    v8f;

// ---------------- WMMA fragment loaders (contiguous k order, same for A and B) ----
// A-matrix 16x32 f16: lane (hi,m=lane&15); element e -> K = hi*16 + e (contiguous 16)
__device__ __forceinline__ v16h frag_a_rowmajor(const _Float16* __restrict__ p,
                                                int ld, int row, int kbase, int lane) {
  int hi = (lane >> 4) & 1;
  const _Float16* rp = p + (size_t)(row + (lane & 15)) * ld + kbase + hi * 16;
  v16h a;
#pragma unroll
  for (int i = 0; i < 16; ++i) a[i] = rp[i];
  return a;
}

// A fragment from Q'/K' view: row n, col f=t*64+dk ; addr = n*TDm + (f>>6)*DDm + (f&63)
__device__ __forceinline__ v16h frag_a_qk(const _Float16* __restrict__ base,
                                          int row, int fbase, int lane) {
  int hi = (lane >> 4) & 1;
  const _Float16* rp = base + (size_t)(row + (lane & 15)) * TDm;
  int f0 = fbase + hi * 16;         // run of 16, never crosses a 64-boundary
  const _Float16* p0 = rp + (f0 >> 6) * DDm + (f0 & 63);
  v16h a;
#pragma unroll
  for (int i = 0; i < 16; ++i) a[i] = p0[i];
  return a;
}

// B-matrix 32x16 f16: lane col = lane&15 ; element e -> K = hi*16 + e (contiguous 16)
__device__ __forceinline__ v16h frag_b_rowmajor(const _Float16* __restrict__ p,
                                                int ld, int col0, int kbase, int lane) {
  int hi = (lane >> 4) & 1;
  const _Float16* rp = p + (size_t)(col0 + (lane & 15)) * ld + kbase + hi * 16;
  v16h a;
#pragma unroll
  for (int i = 0; i < 16; ++i) a[i] = rp[i];
  return a;
}

// B fragment from K' rows (scores: B[k=f][j] = K'[j][f]) : 16 contiguous f per lane-half
__device__ __forceinline__ v16h frag_b_qk(const _Float16* __restrict__ base,
                                          int col0, int fbase, int lane) {
  int hi = (lane >> 4) & 1;
  int f0 = fbase + hi * 16;
  const _Float16* rp = base + (size_t)(col0 + (lane & 15)) * TDm + (f0 >> 6) * DDm + (f0 & 63);
  v16h a;
#pragma unroll
  for (int i = 0; i < 16; ++i) a[i] = rp[i];
  return a;
}

// B fragment from V' (att: B[k=key][f]) : per-lane fixed f column, stride TDm over keys
__device__ __forceinline__ v16h frag_b_vcol(const _Float16* __restrict__ base,
                                            int fcol, int kbase, int lane) {
  int hi = (lane >> 4) & 1;
  int f = fcol + (lane & 15);
  const _Float16* cp = base + (f >> 6) * DDm + (f & 63);
  int k0 = kbase + hi * 16;
  v16h a;
#pragma unroll
  for (int i = 0; i < 16; ++i) a[i] = cp[(size_t)(k0 + i) * TDm];
  return a;
}

// ---------------- async copy: 16B global -> LDS (ASYNCcnt path) ----------------
__device__ __forceinline__ void async_g2l_b128(unsigned lds_off, const void* gaddr) {
  asm volatile("global_load_async_to_lds_b128 %0, %1, off"
               :: "v"(lds_off), "v"(gaddr) : "memory");
}
__device__ __forceinline__ void wait_async0() {
  asm volatile("s_wait_asynccnt 0x0" ::: "memory");
}

// ---------------- kernel 1: (B,D,T,N) -> (B,N,T,D), f32 + f16(+pe) copies ----
__global__ __launch_bounds__(256) void prep_kernel(const float* __restrict__ x,
    const float* __restrict__ pe, float* __restrict__ xr,
    _Float16* __restrict__ xpeh, _Float16* __restrict__ xrh) {
  __shared__ float tile[32][33];
  int bt = blockIdx.z, b = bt >> 5, t = bt & 31;
  int d0 = blockIdx.x * 32, n0 = blockIdx.y * 32;
  int tx = threadIdx.x & 31, ty = threadIdx.x >> 5;
#pragma unroll
  for (int kk = 0; kk < 4; ++kk) {
    int d = d0 + ty + kk * 8;
    tile[ty + kk * 8][tx] = x[(((size_t)b * DDm + d) * TTm + t) * NNm + n0 + tx];
  }
  __syncthreads();
#pragma unroll
  for (int kk = 0; kk < 4; ++kk) {
    int n = n0 + ty + kk * 8;
    int d = d0 + tx;
    float v = tile[tx][ty + kk * 8];
    size_t o = (((size_t)b * NNm + n) * TTm + t) * DDm + d;
    xr[o] = v;
    xrh[o] = (_Float16)v;
    xpeh[o] = (_Float16)(v + pe[(size_t)n * DDm + d]);
  }
}

// ---------------- WMMA GEMM: C[m,o] = A[m,:] . W[o,:] + bias ----------------
// Block: 128 threads = 4 waves. Block tile 128x64; wave tile 32x64 (2x4 accs).
// Weight tile (64 x 32 f16) double-buffered in LDS via async global->LDS DMA.
// EPI 0: f16 store ; 1: f16 relu store ; 2: f32 store + residual
template <int EPI>
__global__ __launch_bounds__(128) void gemm_kernel(const _Float16* __restrict__ A,
    const _Float16* __restrict__ W, const float* __restrict__ bias,
    const float* __restrict__ res, void* __restrict__ out, int K, int O) {
  __shared__ _Float16 wbuf[2][64 * 32];         // 2 x 4 KB
  int tid = threadIdx.x;
  int lane = tid & 31;
  int wave = tid >> 5;
  int m0 = blockIdx.x * 128 + wave * 32;
  int n0 = blockIdx.y * 64;

  unsigned lb0 = (unsigned)(size_t)&wbuf[0][0];
  unsigned lb1 = (unsigned)(size_t)&wbuf[1][0];

  // issue weight tile [n0..n0+63] x [k..k+31] into LDS buffer (2 x 16B per thread)
  auto issue_w = [&](unsigned ldsbase, int k) {
#pragma unroll
    for (int i = 0; i < 2; ++i) {
      int chunk = tid * 2 + i;                  // 0..255
      int row = chunk >> 2;                     // 0..63
      int boff = (chunk & 3) * 16;              // 0/16/32/48 bytes within 64B row
      const char* g = (const char*)(W + (size_t)(n0 + row) * K + k) + boff;
      async_g2l_b128(ldsbase + row * 64 + boff, g);
    }
  };

  v8f acc0[4] = {};
  v8f acc1[4] = {};
  int nk = K >> 5;
  issue_w(lb0, 0);
  for (int kt = 0; kt < nk; ++kt) {
    wait_async0();                              // my async writes for buf[kt&1] done
    __syncthreads();                            // whole tile visible; prev compute done
    if (kt + 1 < nk) issue_w((kt & 1) ? lb0 : lb1, (kt + 1) << 5);
    const _Float16* wl = (const _Float16*)&wbuf[kt & 1][0];
    int k = kt << 5;
    v16h a0 = frag_a_rowmajor(A, K, m0, k, lane);
    v16h a1 = frag_a_rowmajor(A, K, m0 + 16, k, lane);
#pragma unroll
    for (int j = 0; j < 4; ++j) {
      v16h bf = frag_b_rowmajor(wl, 32, 16 * j, 0, lane);
      acc0[j] = __builtin_amdgcn_wmma_f32_16x16x32_f16(false, a0, false, bf,
                                                       (short)0, acc0[j], false, false);
      acc1[j] = __builtin_amdgcn_wmma_f32_16x16x32_f16(false, a1, false, bf,
                                                       (short)0, acc1[j], false, false);
    }
  }

  int hi = (lane >> 4) & 1, cl = lane & 15;
#pragma unroll
  for (int j = 0; j < 4; ++j) {
    int n = n0 + 16 * j + cl;
    float bv = bias[n];
#pragma unroll
    for (int r = 0; r < 8; ++r) {
      float v0 = acc0[j][r] + bv;
      float v1 = acc1[j][r] + bv;
      size_t o0 = (size_t)(m0 + r + 8 * hi) * O + n;
      size_t o1 = (size_t)(m0 + 16 + r + 8 * hi) * O + n;
      if (EPI == 0) {
        ((_Float16*)out)[o0] = (_Float16)v0;
        ((_Float16*)out)[o1] = (_Float16)v1;
      } else if (EPI == 1) {
        ((_Float16*)out)[o0] = (_Float16)(v0 > 0.f ? v0 : 0.f);
        ((_Float16*)out)[o1] = (_Float16)(v1 > 0.f ? v1 : 0.f);
      } else {
        ((float*)out)[o0] = v0 + res[o0];
        ((float*)out)[o1] = v1 + res[o1];
      }
    }
  }
}

// ---------------- attention: per (b,h) scores->softmax->att, all via WMMA ----
__global__ __launch_bounds__(256) void attn_kernel(const _Float16* __restrict__ Qh,
    const _Float16* __restrict__ Kh, const _Float16* __restrict__ Vh,
    const float* __restrict__ adj, const float* __restrict__ hw,
    _Float16* __restrict__ atth) {
  __shared__ float    Sf[64 * 64];     // 16 KB
  __shared__ _Float16 Ahs[64 * 64];    // 8 KB
  int bh = blockIdx.x, b = bh / HHm, h = bh % HHm;
  size_t base = (size_t)b * NTD + (size_t)h * DKm;
  const _Float16* Qb = Qh + base;
  const _Float16* Kb = Kh + base;
  const _Float16* Vb = Vh + base;
  int lane = threadIdx.x & 31;
  int wave = threadIdx.x >> 5;
  int hi = (lane >> 4) & 1, cl = lane & 15;

  // ---- scores S = Q' K'^T * SCALE : 16 tiles of 16x16, K=2048
#pragma unroll
  for (int tt2 = 0; tt2 < 2; ++tt2) {
    int tile = wave * 2 + tt2;
    int mt = (tile >> 2) * 16, nt = (tile & 3) * 16;
    v8f acc = {};
    for (int k = 0; k < FDIM; k += 32) {
      v16h af = frag_a_qk(Qb, mt, k, lane);
      v16h bf = frag_b_qk(Kb, nt, k, lane);
      acc = __builtin_amdgcn_wmma_f32_16x16x32_f16(false, af, false, bf,
                                                   (short)0, acc, false, false);
    }
#pragma unroll
    for (int r = 0; r < 8; ++r)
      Sf[(mt + r + 8 * hi) * 64 + nt + cl] = acc[r] * ATT_SCALE;
  }
  __syncthreads();

  // ---- row softmax, * hw + adj, requantize to f16 (threads 0..63, one row each)
  if (threadIdx.x < 64) {
    int r = threadIdx.x;
    const float* srow = &Sf[r * 64];
    float mx = -1e30f;
    for (int j = 0; j < 64; ++j) mx = fmaxf(mx, srow[j]);
    float s = 0.f;
    for (int j = 0; j < 64; ++j) s += __expf(srow[j] - mx);
    float inv = 1.0f / s;
    float hwv = hw[h];
    const float* arow = adj + ((size_t)h * NNm + r) * NNm;
    for (int j = 0; j < 64; ++j)
      Ahs[r * 64 + j] = (_Float16)(__expf(srow[j] - mx) * inv * hwv + arow[j]);
  }
  __syncthreads();

  // ---- att = A(64x64) . V'(64x2048), 512 tiles of 16x16, K=64
  for (int tile = wave; tile < 4 * 128; tile += 8) {
    int mt = (tile >> 7) * 16;
    int ft = (tile & 127) * 16;
    v8f acc = {};
#pragma unroll
    for (int k = 0; k < 64; k += 32) {
      v16h af = frag_a_rowmajor((const _Float16*)Ahs, 64, mt, k, lane);
      v16h bf = frag_b_vcol(Vb, ft, k, lane);
      acc = __builtin_amdgcn_wmma_f32_16x16x32_f16(false, af, false, bf,
                                                   (short)0, acc, false, false);
    }
    int f = ft + cl;                                  // 16 cols share one t-block
    _Float16* op = atth + base + (f >> 6) * DDm + (f & 63);
#pragma unroll
    for (int r = 0; r < 8; ++r)
      op[(size_t)(mt + r + 8 * hi) * TDm] = (_Float16)acc[r];
  }
}

// ---------------- layernorm: pass 1 partial sums ----------------
__global__ __launch_bounds__(256) void lnred_kernel(const float* __restrict__ xp,
                                                    float* __restrict__ red) {
  int b = blockIdx.x;
  const float* p = xp + (size_t)b * NTD + (size_t)blockIdx.y * (NTD / 64);
  float s = 0.f, s2 = 0.f;
  for (int i = threadIdx.x; i < NTD / 64; i += 256) {
    float v = p[i]; s += v; s2 += v * v;
  }
  __shared__ float sh0[256], sh1[256];
  sh0[threadIdx.x] = s; sh1[threadIdx.x] = s2;
  __syncthreads();
  for (int st = 128; st > 0; st >>= 1) {
    if (threadIdx.x < st) { sh0[threadIdx.x] += sh0[threadIdx.x + st];
                            sh1[threadIdx.x] += sh1[threadIdx.x + st]; }
    __syncthreads();
  }
  if (threadIdx.x == 0) { atomicAdd(&red[2 * b], sh0[0]); atomicAdd(&red[2 * b + 1], sh1[0]); }
}

// ---------------- layernorm: pass 2 apply (in-place f32 + f16 copy) ----------
__global__ __launch_bounds__(256) void lnapp_kernel(float* __restrict__ x1,
    _Float16* __restrict__ x1h, const float* __restrict__ red,
    const float* __restrict__ g, const float* __restrict__ be) {
  size_t idx = (size_t)blockIdx.x * 256 + threadIdx.x;
  int b = (int)(idx >> 20);          // NTD == 2^20
  int r = (int)(idx & (NTD - 1));
  const float inv = 1.0f / (float)NTD;
  float mean = red[2 * b] * inv;
  float var  = red[2 * b + 1] * inv - mean * mean;
  float rs = rsqrtf(var + LN_EPS);
  float v = (x1[idx] - mean) * rs * g[r] + be[r];
  x1[idx] = v;
  x1h[idx] = (_Float16)v;
}

// ---------------- kernel 9: (B,N,T,D) -> (B,D,T,N) ----------------
__global__ __launch_bounds__(256) void outt_kernel(const float* __restrict__ pre,
                                                   float* __restrict__ out) {
  __shared__ float tile[32][33];
  int bt = blockIdx.z, b = bt >> 5, t = bt & 31;
  int d0 = blockIdx.x * 32, n0 = blockIdx.y * 32;
  int tx = threadIdx.x & 31, ty = threadIdx.x >> 5;
#pragma unroll
  for (int kk = 0; kk < 4; ++kk) {
    int n = n0 + ty + kk * 8;
    tile[ty + kk * 8][tx] = pre[(((size_t)b * NNm + n) * TTm + t) * DDm + d0 + tx];
  }
  __syncthreads();
#pragma unroll
  for (int kk = 0; kk < 4; ++kk) {
    int d = d0 + ty + kk * 8;
    out[(((size_t)b * DDm + d) * TTm + t) * NNm + n0 + tx] = tile[tx][ty + kk * 8];
  }
}

// ---------------- f32 -> f16 weight conversion ----------------
__global__ void cvt_kernel(const float* __restrict__ s, _Float16* __restrict__ d, int n) {
  int i = blockIdx.x * 256 + threadIdx.x;
  if (i < n) d[i] = (_Float16)s[i];
}

// ---------------- host orchestration ----------------
extern "C" void kernel_launch(void* const* d_in, const int* in_sizes, int n_in,
                              void* d_out, int out_size, void* d_ws, size_t ws_size,
                              hipStream_t stream) {
  (void)in_sizes; (void)n_in; (void)out_size;
  const float* x   = (const float*)d_in[0];
  const float* Wq  = (const float*)d_in[1];  const float* bq = (const float*)d_in[2];
  const float* Wk  = (const float*)d_in[3];  const float* bk = (const float*)d_in[4];
  const float* Wv  = (const float*)d_in[5];  const float* bv = (const float*)d_in[6];
  const float* Wo  = (const float*)d_in[7];  const float* bo = (const float*)d_in[8];
  const float* W1  = (const float*)d_in[9];  const float* b1 = (const float*)d_in[10];
  const float* W2  = (const float*)d_in[11]; const float* b2 = (const float*)d_in[12];
  const float* g1  = (const float*)d_in[13]; const float* be1 = (const float*)d_in[14];
  const float* adj = (const float*)d_in[15]; const float* hw = (const float*)d_in[16];
  const float* pe  = (const float*)d_in[17];

  char* ws = (char*)d_ws;
  const size_t S32 = (size_t)MROWS * DDm * 4;   // 134,217,728 B
  const size_t S16 = S32 / 2;                   //  67,108,864 B
  const size_t WEL = (size_t)DDm * DDm;         // 262,144 elements per square weight
  const size_t NEED = 2 * S32 + 6 * S16 + 1024 + 12 * WEL * 2;
  if (ws_size < NEED) return;                   // deterministic no-op guard

  float*    xr   = (float*)ws;                  // also reused as pre-output buffer
  float*    x1   = (float*)(ws + S32);
  _Float16* xpeh = (_Float16*)(ws + 2 * S32);
  _Float16* xrh  = (_Float16*)(ws + 2 * S32 + S16);
  _Float16* Qh   = (_Float16*)(ws + 2 * S32 + 2 * S16);
  _Float16* Khb  = (_Float16*)(ws + 2 * S32 + 3 * S16);
  _Float16* ffh  = (_Float16*)(ws + 2 * S32);             // overlays xpeh..Khb (4*S16)
  _Float16* Vh   = (_Float16*)(ws + 2 * S32 + 4 * S16);
  _Float16* x1h  = Vh;                                    // overlays Vh
  _Float16* atth = (_Float16*)(ws + 2 * S32 + 5 * S16);
  float*    red  = (float*)(ws + 2 * S32 + 6 * S16);
  _Float16* Wqh  = (_Float16*)(ws + 2 * S32 + 6 * S16 + 1024);
  _Float16* Wkh = Wqh + WEL;  _Float16* Wvh = Wkh + WEL;  _Float16* Woh = Wvh + WEL;
  _Float16* W1h = Woh + WEL;  _Float16* W2h = W1h + 4 * WEL;

  // weight conversion (tiny)
  cvt_kernel<<<dim3((int)((WEL + 255) / 256)), 256, 0, stream>>>(Wq, Wqh, (int)WEL);
  cvt_kernel<<<dim3((int)((WEL + 255) / 256)), 256, 0, stream>>>(Wk, Wkh, (int)WEL);
  cvt_kernel<<<dim3((int)((WEL + 255) / 256)), 256, 0, stream>>>(Wv, Wvh, (int)WEL);
  cvt_kernel<<<dim3((int)((WEL + 255) / 256)), 256, 0, stream>>>(Wo, Woh, (int)WEL);
  cvt_kernel<<<dim3((int)((4 * WEL + 255) / 256)), 256, 0, stream>>>(W1, W1h, (int)(4 * WEL));
  cvt_kernel<<<dim3((int)((4 * WEL + 255) / 256)), 256, 0, stream>>>(W2, W2h, (int)(4 * WEL));

  // 1) transpose in + pe
  prep_kernel<<<dim3(DDm / 32, NNm / 32, BB * TTm), 256, 0, stream>>>(x, pe, xr, xpeh, xrh);

  // 2) Q, K, V projections (WMMA, async-LDS weight staging)
  dim3 gproj(MROWS / 128, DDm / 64);
  gemm_kernel<0><<<gproj, 128, 0, stream>>>(xpeh, Wqh, bq, nullptr, Qh, DDm, DDm);
  gemm_kernel<0><<<gproj, 128, 0, stream>>>(xpeh, Wkh, bk, nullptr, Khb, DDm, DDm);
  gemm_kernel<0><<<gproj, 128, 0, stream>>>(xrh,  Wvh, bv, nullptr, Vh, DDm, DDm);

  // 3) attention per (b,h)
  attn_kernel<<<BB * HHm, 256, 0, stream>>>(Qh, Khb, Vh, adj, hw, atth);

  // 4) output projection + residual -> x1pre (f32)
  gemm_kernel<2><<<gproj, 128, 0, stream>>>(atth, Woh, bo, xr, x1, DDm, DDm);

  // 5) layernorm over (N,T,D) per batch
  (void)hipMemsetAsync(red, 0, BB * 2 * sizeof(float), stream);
  lnred_kernel<<<dim3(BB, 64), 256, 0, stream>>>(x1, red);
  lnapp_kernel<<<dim3((int)(((size_t)MROWS * DDm) / 256)), 256, 0, stream>>>(x1, x1h, red, g1, be1);

  // 6) FFN: relu(x1 W1^T + b1) W2^T + b2 + x1
  gemm_kernel<1><<<dim3(MROWS / 128, (4 * DDm) / 64), 128, 0, stream>>>(x1h, W1h, b1, nullptr, ffh, DDm, 4 * DDm);
  gemm_kernel<2><<<gproj, 128, 0, stream>>>(ffh, W2h, b2, x1, xr, 4 * DDm, DDm);

  // 7) transpose out
  outt_kernel<<<dim3(DDm / 32, NNm / 32, BB * TTm), 256, 0, stream>>>(xr, (float*)d_out);
}
